// sinc_conv_14645838480142
// MI455X (gfx1250) — matrix operand order repaired and checked by reading, not compiled, not run on hardware
//
#include <hip/hip_runtime.h>

// SincNet conv for MI455X (gfx1250): filter-bank build + implicit-GEMM conv
// using v_wmma_f32_16x16x32_f16 (f16 inputs, f32 accumulate).
//
// GEMM view: D[f, p] = sum_k filt[f,k] * x[p+k]
//   M = 32 filters (2 tiles of 16), K = 101 padded to 128 (4 steps of 32),
//   N = 32*64*1900 output positions.

typedef __attribute__((ext_vector_type(16))) _Float16 v16h;
typedef __attribute__((ext_vector_type(8)))  float    v8f;

#define N_FILT   32
#define FILT_DIM 101
#define KP       128          // K padded to 4 WMMA steps of 32
#define W_IN     2000
#define W_OUT    1900
#define H_DIM    64
#define N_BATCH  32
#define CHUNK    512          // output positions per workgroup
#define SPAN     640          // CHUNK + KP (padded x span staged in LDS)
#define FSTRIDE  (H_DIM * W_OUT)   // per-filter stride in output (121600)

// ---------------------------------------------------------------------------
// Stage 1: build 32x128 f16 filter bank in workspace (zero-padded K to 128).
// band_pass[k] = (sin(2*pi*end*d) - sin(2*pi*beg*d)) / (pi*d), d = |k-50|
//              = 2*(end-beg) at d==0; then /max over k, then Hamming window.
// One block per filter; 128 threads; LDS tree-reduce for the max.
// ---------------------------------------------------------------------------
__global__ __launch_bounds__(128)
void sincnet_build_filters(const float* __restrict__ filt_b1,
                           const float* __restrict__ filt_band,
                           _Float16* __restrict__ filt16) {
    __shared__ float red[128];
    const int f = blockIdx.x;
    const int k = threadIdx.x;              // 0..127

    const float beg = fabsf(filt_b1[f]) + 1.0f / 250.0f;
    const float end = beg + fabsf(filt_band[f]) + 5.0f / 250.0f;

    float bp = 0.0f;
    if (k < FILT_DIM) {
        int d = k - 50; if (d < 0) d = -d;
        if (d == 0) {
            bp = 2.0f * (end - beg);
        } else {
            const float pd = 3.14159265358979323846f * (float)d;
            bp = (sinf(2.0f * pd * end) - sinf(2.0f * pd * beg)) / pd;
        }
    }
    red[k] = (k < FILT_DIM) ? bp : -1e30f;
    __syncthreads();
    for (int s = 64; s > 0; s >>= 1) {
        if (k < s) red[k] = fmaxf(red[k], red[k + s]);
        __syncthreads();
    }
    const float mx = red[0];

    float val = 0.0f;
    if (k < FILT_DIM) {
        // window[k] = 0.54 - 0.46*cos(2*pi*k/100)  (linspace(0,101,101) step folds to k/100)
        const float win = 0.54f - 0.46f * cosf(6.283185307179586f * (float)k / 100.0f);
        val = (bp / mx) * win;
    }
    filt16[f * KP + k] = (_Float16)val;     // k >= 101 -> zero pad
}

// ---------------------------------------------------------------------------
// Stage 2: WMMA conv. One block = 8 waves, one (batch,h) row x 512 positions.
// LDS holds pre-packed f16 pairs: y[i] = {x16[i], x16[i+1]} so every B-matrix
// fragment VGPR is a single aligned ds_load_b32 (conversion f32->f16 done in
// the fill pass). A fragments load as aligned dwords from the 8 KB filter
// table (L2-resident).
// ---------------------------------------------------------------------------
__global__ __launch_bounds__(256)
void sincnet_conv_wmma(const float* __restrict__ x,
                       const _Float16* __restrict__ filt16,
                       float* __restrict__ out) {
    __shared__ unsigned int ylds[SPAN];

    const int blk   = blockIdx.x;
    const int row   = blk >> 2;             // 0..2047 = batch*64 + h
    const int chunk = blk & 3;
    const int nb    = row >> 6;             // batch index
    const int h     = row & 63;
    const int pbase = chunk * CHUNK;
    const float* xrow = x + (size_t)row * W_IN;

    // Fill packed-pair f16 LDS image of x[pbase .. pbase+SPAN] (clamped to 0 OOB)
    for (int i = threadIdx.x; i < SPAN; i += 256) {
        const int ia = pbase + i;
        const int ib = ia + 1;
        const float fa = (ia < W_IN) ? xrow[ia] : 0.0f;
        const float fb = (ib < W_IN) ? xrow[ib] : 0.0f;
        const _Float16 ha = (_Float16)fa, hb = (_Float16)fb;
        const unsigned int ua = __builtin_bit_cast(unsigned short, ha);
        const unsigned int ub = __builtin_bit_cast(unsigned short, hb);
        ylds[i] = ua | (ub << 16);
    }
    __syncthreads();

    const int lane   = threadIdx.x & 31;
    const int wave   = threadIdx.x >> 5;
    const int m      = lane & 15;           // M row (A) / N col (B,D)
    const int hiHalf = lane >> 4;           // lane 16..31 flag

    typedef union { v16h h; unsigned int u[8]; } frag;

    // Preload A fragments [mtile][kstep]; ISA 16-bit A layout:
    //   vgpr j: K = (j>=4 ? 16:0) + (lane>=16 ? 8:0) + 2*(j&3) (+0/1 in halves)
    // All offsets even -> each fragment dword is one aligned 32-bit load.
    frag A[2][4];
#pragma unroll
    for (int mt = 0; mt < 2; ++mt) {
        const int fbase = (mt * 16 + m) * KP + hiHalf * 8;
#pragma unroll
        for (int s = 0; s < 4; ++s) {
#pragma unroll
            for (int j = 0; j < 8; ++j) {
                const int off = fbase + 32 * s + ((j >= 4) ? 16 : 0) + 2 * (j & 3);
                A[mt][s].u[j] = *(const unsigned int*)(filt16 + off);
            }
        }
    }

    const size_t outbase = (size_t)nb * (size_t)(N_FILT * FSTRIDE) + (size_t)h * W_OUT;

#pragma unroll
    for (int t = 0; t < 4; ++t) {
        const int poff = wave * 64 + t * 16;    // tile position offset in chunk
        v8f acc0 = {};                          // filters 0..15
        v8f acc1 = {};                          // filters 16..31
#pragma unroll
        for (int s = 0; s < 4; ++s) {
            // B layout (32x16, 16-bit): vgpr j, lane L: N=L&15,
            //   K = (j>=4 ? 16:0) + (L>=16 ? 8:0) + 2*(j&3) (+0/1) -> packed pair = ylds[a0+2j+(j>=4?8:0)]
            frag B;
            const int a0 = poff + m + 32 * s + hiHalf * 8;
#pragma unroll
            for (int j = 0; j < 8; ++j) B.u[j] = ylds[a0 + 2 * j + ((j >= 4) ? 8 : 0)];

            acc0 = __builtin_amdgcn_wmma_f32_16x16x32_f16(
                false, A[0][s].h, false, B.h, (short)0, acc0, false, false);
            acc1 = __builtin_amdgcn_wmma_f32_16x16x32_f16(
                false, A[1][s].h, false, B.h, (short)0, acc1, false, false);
        }

        // D layout: vgpr r -> filter row r + 8*hiHalf; lane N = m -> w
        const int w = pbase + poff + m;
        if (w < W_OUT) {
#pragma unroll
            for (int r = 0; r < 8; ++r) {
                const int f0 = r + 8 * hiHalf;
                out[outbase + (size_t)f0        * FSTRIDE + w] = acc0[r];
                out[outbase + (size_t)(f0 + 16) * FSTRIDE + w] = acc1[r];
            }
        }
    }
}

// ---------------------------------------------------------------------------
extern "C" void kernel_launch(void* const* d_in, const int* in_sizes, int n_in,
                              void* d_out, int out_size, void* d_ws, size_t ws_size,
                              hipStream_t stream) {
    const float* x    = (const float*)d_in[0];   // (32,1,64,2000) f32
    const float* b1   = (const float*)d_in[1];   // (32,) f32
    const float* band = (const float*)d_in[2];   // (32,) f32
    float* out        = (float*)d_out;           // (32,32,64,1900) f32
    _Float16* filt16  = (_Float16*)d_ws;         // 32*128 f16 = 8 KB scratch

    sincnet_build_filters<<<N_FILT, 128, 0, stream>>>(b1, band, filt16);

    const int rows = N_BATCH * H_DIM;            // 2048
    const int chunks = (W_OUT + CHUNK - 1) / CHUNK;  // 4
    sincnet_conv_wmma<<<rows * chunks, 256, 0, stream>>>(x, filt16, out);
}
